// MultiSingleStreamBlockLoraProcessor_20160576487592
// MI455X (gfx1250) — compile-verified, matrix operand order checked by reading, not run on hardware
//
#include <hip/hip_runtime.h>
#include <hip/hip_bf16.h>
#include <cstdint>
#include <cstddef>

#define S_LEN  2048
#define SC_LEN 1024
#define DMODEL 3072
#define NHEADS 24
#define HDIM   128

typedef __attribute__((ext_vector_type(16))) _Float16 v16h;
typedef __attribute__((ext_vector_type(8)))  float    v8f;
typedef __attribute__((ext_vector_type(4)))  _Float16 v4h;

union AF { uint4 u[2]; v16h h; };

static __device__ __forceinline__ v8f wmma_f16(v16h a, v16h b, v8f c) {
  // D = A(16x32 f16) x B(32x16 f16) + C(16x16 f32)
  return __builtin_amdgcn_wmma_f32_16x16x32_f16(false, a, false, b, (short)0, c,
                                                false, false);
}

static __device__ __forceinline__ v8f v8f_zero() {
  v8f z = {0.f, 0.f, 0.f, 0.f, 0.f, 0.f, 0.f, 0.f};
  return z;
}

// ---------------------------------------------------------------------------
// fp32 -> f16 elementwise convert
// ---------------------------------------------------------------------------
__global__ void f32_to_f16_kernel(const float* __restrict__ in,
                                  _Float16* __restrict__ out, size_t n) {
  size_t i = (size_t)blockIdx.x * blockDim.x + threadIdx.x;
  if (i < n) out[i] = (_Float16)in[i];
}

// ---------------------------------------------------------------------------
// C[M,N] = X[M,K] * W[N,K]^T + bias[N]   (f16 inputs, f32 out)
// block = 128 threads (4 waves); block tile M=64 x N=128.
// per wave: 32x64 tile -> 8 WMMA per 12 b128 loads per 32-k step.
// ---------------------------------------------------------------------------
__global__ void __launch_bounds__(128)
gemm_wmma_kernel(const _Float16* __restrict__ X, const _Float16* __restrict__ W,
                 const float* __restrict__ bias, float* __restrict__ out,
                 int M, int N, int K) {
  const int lane = threadIdx.x & 31;
  const int wave = threadIdx.x >> 5;
  const int r    = lane & 15;
  const int half = lane >> 4;
  const int m0   = blockIdx.y * 64 + (wave >> 1) * 32;
  const int n0   = blockIdx.x * 128 + (wave & 1) * 64;

  v8f acc[2][4];
#pragma unroll
  for (int a = 0; a < 2; ++a)
#pragma unroll
    for (int j = 0; j < 4; ++j) acc[a][j] = v8f_zero();

  const _Float16* xr0 = X + (size_t)(m0 + r) * K;
  const _Float16* xr1 = xr0 + (size_t)16 * K;

  for (int k = 0; k < K; k += 32) {
    AF a0, a1;
    a0.u[0] = *(const uint4*)(xr0 + k + half * 8);
    a0.u[1] = *(const uint4*)(xr0 + k + 16 + half * 8);
    a1.u[0] = *(const uint4*)(xr1 + k + half * 8);
    a1.u[1] = *(const uint4*)(xr1 + k + 16 + half * 8);
#pragma unroll
    for (int j = 0; j < 4; ++j) {
      const _Float16* wr = W + (size_t)(n0 + j * 16 + r) * K + k;
      AF b;
      b.u[0] = *(const uint4*)(wr + half * 8);
      b.u[1] = *(const uint4*)(wr + 16 + half * 8);
      acc[0][j] = wmma_f16(a0.h, b.h, acc[0][j]);
      acc[1][j] = wmma_f16(a1.h, b.h, acc[1][j]);
    }
  }

#pragma unroll
  for (int a = 0; a < 2; ++a) {
#pragma unroll
    for (int j = 0; j < 4; ++j) {
      const int n = n0 + j * 16 + r;
      const float bn = bias[n];
#pragma unroll
      for (int i = 0; i < 8; ++i) {
        const int m = m0 + a * 16 + half * 8 + i;
        out[(size_t)m * N + n] = acc[a][j][i] + bn;
      }
    }
  }
}

// ---------------------------------------------------------------------------
// RMSNorm(head_dim) + RoPE + optional scale, fp32 [T,D] -> f16 [H][T][HD]
// one wave per (h,t) row; lane owns 4 consecutive dims (2 rope pairs)
// ---------------------------------------------------------------------------
__global__ void __launch_bounds__(128)
norm_rope_pack_kernel(const float* __restrict__ P, const float* __restrict__ w,
                      const float* __restrict__ cs, const float* __restrict__ sn,
                      _Float16* __restrict__ out, int T, float scale) {
  const int gw   = (int)((blockIdx.x * blockDim.x + threadIdx.x) >> 5);
  const int lane = threadIdx.x & 31;
  if (gw >= NHEADS * T) return;
  const int t = gw % T;
  const int h = gw / T;
  const int d = lane * 4;

  const float4 x = *(const float4*)(P + (size_t)t * DMODEL + h * HDIM + d);
  float ss = x.x * x.x + x.y * x.y + x.z * x.z + x.w * x.w;
#pragma unroll
  for (int msk = 1; msk < 32; msk <<= 1) ss += __shfl_xor(ss, msk, 32);
  const float inv = rsqrtf(ss * (1.0f / (float)HDIM) + 1e-6f);

  const float4 wv = *(const float4*)(w + d);
  const float4 c  = *(const float4*)(cs + (size_t)t * HDIM + d);
  const float4 s  = *(const float4*)(sn + (size_t)t * HDIM + d);

  const float y0 = x.x * inv * wv.x;
  const float y1 = x.y * inv * wv.y;
  const float y2 = x.z * inv * wv.z;
  const float y3 = x.w * inv * wv.w;

  v4h o;
  o[0] = (_Float16)((y0 * c.x - y1 * s.x) * scale);
  o[1] = (_Float16)((y1 * c.y + y0 * s.y) * scale);
  o[2] = (_Float16)((y2 * c.z - y3 * s.z) * scale);
  o[3] = (_Float16)((y3 * c.w + y2 * s.w) * scale);
  *(v4h*)(out + ((size_t)h * T + t) * HDIM + d) = o;
}

// ---------------------------------------------------------------------------
// V pack + transpose: fp32 [T,D] -> f16 [H][HD][T]  (out[hd*T + t])
// LDS-tiled 32x32 so both global read and write are coalesced.
// block = 256 threads; grid = (DMODEL/32, T/32)
// ---------------------------------------------------------------------------
__global__ void __launch_bounds__(256)
vtranspose_pack_kernel(const float* __restrict__ P, _Float16* __restrict__ out,
                       int T) {
  __shared__ _Float16 tile[32][33];  // +1 pad to dodge bank conflicts
  const int tx = threadIdx.x & 31;
  const int ty = threadIdx.x >> 5;   // 0..7
  const int d0 = blockIdx.x * 32;
  const int t0 = blockIdx.y * 32;
#pragma unroll
  for (int k = 0; k < 4; ++k) {
    const int t = t0 + ty + k * 8;
    tile[ty + k * 8][tx] = (_Float16)P[(size_t)t * DMODEL + d0 + tx];
  }
  __syncthreads();
#pragma unroll
  for (int k = 0; k < 4; ++k) {
    const int row = ty + k * 8;
    out[(size_t)(d0 + row) * T + t0 + tx] = tile[tx][row];
  }
}

// ---------------------------------------------------------------------------
// One online-softmax attention pass over T keys for a 16-query block.
// ---------------------------------------------------------------------------
__device__ __forceinline__ void attn_pass(const AF* qa,
                                          const _Float16* __restrict__ Kr,
                                          const _Float16* __restrict__ Vt,
                                          int T, _Float16* __restrict__ ldsP,
                                          int lane, v8f* o, float* m, float* l) {
  const int r = lane & 15, half = lane >> 4;
#pragma unroll
  for (int j = 0; j < 8; ++j) { o[j] = v8f_zero(); }
#pragma unroll
  for (int i = 0; i < 8; ++i) { m[i] = -1e30f; l[i] = 0.f; }

  for (int k0 = 0; k0 < T; k0 += 32) {
    const _Float16* kr0 = Kr + (size_t)(k0 + r) * HDIM;
    const _Float16* kr1 = kr0 + 16 * HDIM;
    __builtin_prefetch(kr0 + 32 * HDIM, 0, 0);  // global_prefetch_b8, next chunk
    v8f s0 = v8f_zero(), s1 = v8f_zero();
#pragma unroll
    for (int kc = 0; kc < 4; ++kc) {
      AF b0, b1;
      b0.u[0] = *(const uint4*)(kr0 + kc * 32 + half * 8);
      b0.u[1] = *(const uint4*)(kr0 + kc * 32 + 16 + half * 8);
      b1.u[0] = *(const uint4*)(kr1 + kc * 32 + half * 8);
      b1.u[1] = *(const uint4*)(kr1 + kc * 32 + 16 + half * 8);
      s0 = wmma_f16(qa[kc].h, b0.h, s0);
      s1 = wmma_f16(qa[kc].h, b1.h, s1);
    }
    // online softmax: rows live in VGPR index (row = half*8+i), keys across 16 lanes
#pragma unroll
    for (int i = 0; i < 8; ++i) {
      float cm = fmaxf(s0[i], s1[i]);
      cm = fmaxf(cm, __shfl_xor(cm, 1, 32));
      cm = fmaxf(cm, __shfl_xor(cm, 2, 32));
      cm = fmaxf(cm, __shfl_xor(cm, 4, 32));
      cm = fmaxf(cm, __shfl_xor(cm, 8, 32));
      const float mn = fmaxf(m[i], cm);
      const float alpha = __expf(m[i] - mn);
      m[i] = mn;
      const float p0 = __expf(s0[i] - mn);
      const float p1 = __expf(s1[i] - mn);
      s0[i] = p0; s1[i] = p1;
      float rs = p0 + p1;
      rs += __shfl_xor(rs, 1, 32);
      rs += __shfl_xor(rs, 2, 32);
      rs += __shfl_xor(rs, 4, 32);
      rs += __shfl_xor(rs, 8, 32);
      l[i] = l[i] * alpha + rs;
#pragma unroll
      for (int j = 0; j < 8; ++j) o[j][i] *= alpha;
    }
    // transpose P (C-layout) -> A-layout via LDS, converting to f16
#pragma unroll
    for (int i = 0; i < 8; ++i) {
      const int row = half * 8 + i;
      ldsP[row * 32 + r]      = (_Float16)s0[i];
      ldsP[row * 32 + 16 + r] = (_Float16)s1[i];
    }
    AF pa;  // same-wave LDS ops are in-order; no barrier needed
    pa.u[0] = *(const uint4*)(ldsP + r * 32 + half * 8);
    pa.u[1] = *(const uint4*)(ldsP + r * 32 + 16 + half * 8);
#pragma unroll
    for (int j = 0; j < 8; ++j) {
      const _Float16* vr = Vt + (size_t)(j * 16 + r) * T + k0;
      AF bv;
      bv.u[0] = *(const uint4*)(vr + half * 8);
      bv.u[1] = *(const uint4*)(vr + 16 + half * 8);
      o[j] = wmma_f16(pa.h, bv.h, o[j]);
    }
  }
}

// ---------------------------------------------------------------------------
// Fused dual attention: out = lambda * softmax(Q Kc^T) Vc + softmax(Q K^T) V
// 4 waves per block; each wave owns one (head, 16-query) block.
// ---------------------------------------------------------------------------
__global__ void __launch_bounds__(128)
attention_kernel(const _Float16* __restrict__ Qb, const _Float16* __restrict__ Kb,
                 const _Float16* __restrict__ Vtb, const _Float16* __restrict__ Kcb,
                 const _Float16* __restrict__ Vctb, const float* __restrict__ lam,
                 float* __restrict__ out) {
  __shared__ _Float16 lds[4][16 * 32];
  const int wave = threadIdx.x >> 5;
  const int lane = threadIdx.x & 31;
  const int r = lane & 15, half = lane >> 4;
  const int qb = blockIdx.x * 4 + wave;       // 0 .. H*(S/16)-1
  const int h  = qb >> 7;                     // / (S_LEN/16)
  const int q0 = (qb & 127) * 16;

  // Q fragments (pre-scaled by 1/sqrt(HD) at pack time)
  AF qa[4];
  const _Float16* qrow = Qb + ((size_t)h * S_LEN + q0 + r) * HDIM;
#pragma unroll
  for (int kc = 0; kc < 4; ++kc) {
    qa[kc].u[0] = *(const uint4*)(qrow + kc * 32 + half * 8);
    qa[kc].u[1] = *(const uint4*)(qrow + kc * 32 + 16 + half * 8);
  }

  v8f oc[8]; float mc[8], lc[8];
  attn_pass(qa, Kcb + (size_t)h * SC_LEN * HDIM, Vctb + (size_t)h * HDIM * SC_LEN,
            SC_LEN, lds[wave], lane, oc, mc, lc);
  const float lambda = lam[0];
#pragma unroll
  for (int i = 0; i < 8; ++i) {
    const float f = lambda / lc[i];
#pragma unroll
    for (int j = 0; j < 8; ++j) oc[j][i] *= f;
  }

  v8f os[8]; float ms[8], ls[8];
  attn_pass(qa, Kb + (size_t)h * S_LEN * HDIM, Vtb + (size_t)h * HDIM * S_LEN,
            S_LEN, lds[wave], lane, os, ms, ls);

#pragma unroll
  for (int i = 0; i < 8; ++i) {
    const float inv = 1.0f / ls[i];
    const size_t row = (size_t)(q0 + half * 8 + i) * DMODEL + h * HDIM;
#pragma unroll
    for (int j = 0; j < 8; ++j) {
      out[row + j * 16 + r] = oc[j][i] + os[j][i] * inv;
    }
  }
}

// ---------------------------------------------------------------------------
// host launcher
// ---------------------------------------------------------------------------
extern "C" void kernel_launch(void* const* d_in, const int* in_sizes, int n_in,
                              void* d_out, int out_size, void* d_ws, size_t ws_size,
                              hipStream_t stream) {
  (void)in_sizes; (void)n_in; (void)out_size; (void)ws_size;
  const float* hs    = (const float*)d_in[0];
  const float* cond  = (const float*)d_in[1];
  const float* lam   = (const float*)d_in[2];
  const float* Wq    = (const float*)d_in[3];
  const float* bq    = (const float*)d_in[4];
  const float* Wk    = (const float*)d_in[5];
  const float* bk    = (const float*)d_in[6];
  const float* Wv    = (const float*)d_in[7];
  const float* bv    = (const float*)d_in[8];
  const float* nqw   = (const float*)d_in[9];
  const float* nkw   = (const float*)d_in[10];
  const float* cos0  = (const float*)d_in[11];
  const float* sin0  = (const float*)d_in[12];
  const float* cos1  = (const float*)d_in[13];
  const float* sin1  = (const float*)d_in[14];
  float* out = (float*)d_out;

  // workspace carve (256B aligned)
  char* wp = (char*)d_ws;
  auto carve = [&](size_t bytes) -> void* {
    void* p = (void*)wp;
    wp += (bytes + 255) & ~(size_t)255;
    return p;
  };
  _Float16* hsb   = (_Float16*)carve((size_t)S_LEN * DMODEL * 2);
  _Float16* condb = (_Float16*)carve((size_t)SC_LEN * DMODEL * 2);
  _Float16* wbuf  = (_Float16*)carve((size_t)DMODEL * DMODEL * 2);
  float*    pbuf  = (float*)   carve((size_t)S_LEN * DMODEL * 4);
  _Float16* Qb    = (_Float16*)carve((size_t)S_LEN * DMODEL * 2);
  _Float16* Kb    = (_Float16*)carve((size_t)S_LEN * DMODEL * 2);
  _Float16* Kcb   = (_Float16*)carve((size_t)SC_LEN * DMODEL * 2);
  _Float16* Vtb   = (_Float16*)carve((size_t)S_LEN * DMODEL * 2);
  _Float16* Vctb  = (_Float16*)carve((size_t)SC_LEN * DMODEL * 2);

  const float qscale = 0.088388347648318447f;  // 1/sqrt(128)
  const size_t nHS = (size_t)S_LEN * DMODEL;
  const size_t nC  = (size_t)SC_LEN * DMODEL;
  const size_t nW  = (size_t)DMODEL * DMODEL;
  auto cgrid = [](size_t n) { return (unsigned)((n + 255) / 256); };

  f32_to_f16_kernel<<<cgrid(nHS), 256, 0, stream>>>(hs, hsb, nHS);
  f32_to_f16_kernel<<<cgrid(nC), 256, 0, stream>>>(cond, condb, nC);

  const dim3 gS(DMODEL / 128, S_LEN / 64);
  const dim3 gC(DMODEL / 128, SC_LEN / 64);
  const unsigned packS = (unsigned)(NHEADS * S_LEN / 4);
  const unsigned packC = (unsigned)(NHEADS * SC_LEN / 4);
  const dim3 vtS(DMODEL / 32, S_LEN / 32);
  const dim3 vtC(DMODEL / 32, SC_LEN / 32);

  // Q
  f32_to_f16_kernel<<<cgrid(nW), 256, 0, stream>>>(Wq, wbuf, nW);
  gemm_wmma_kernel<<<gS, 128, 0, stream>>>(hsb, wbuf, bq, pbuf, S_LEN, DMODEL, DMODEL);
  norm_rope_pack_kernel<<<packS, 128, 0, stream>>>(pbuf, nqw, cos0, sin0, Qb, S_LEN, qscale);

  // K (self) and Kc (cross)
  f32_to_f16_kernel<<<cgrid(nW), 256, 0, stream>>>(Wk, wbuf, nW);
  gemm_wmma_kernel<<<gS, 128, 0, stream>>>(hsb, wbuf, bk, pbuf, S_LEN, DMODEL, DMODEL);
  norm_rope_pack_kernel<<<packS, 128, 0, stream>>>(pbuf, nkw, cos0, sin0, Kb, S_LEN, 1.0f);
  gemm_wmma_kernel<<<gC, 128, 0, stream>>>(condb, wbuf, bk, pbuf, SC_LEN, DMODEL, DMODEL);
  norm_rope_pack_kernel<<<packC, 128, 0, stream>>>(pbuf, nkw, cos1, sin1, Kcb, SC_LEN, 1.0f);

  // V (self) and Vc (cross), transposed to [H][HD][T]
  f32_to_f16_kernel<<<cgrid(nW), 256, 0, stream>>>(Wv, wbuf, nW);
  gemm_wmma_kernel<<<gS, 128, 0, stream>>>(hsb, wbuf, bv, pbuf, S_LEN, DMODEL, DMODEL);
  vtranspose_pack_kernel<<<vtS, 256, 0, stream>>>(pbuf, Vtb, S_LEN);
  gemm_wmma_kernel<<<gC, 128, 0, stream>>>(condb, wbuf, bv, pbuf, SC_LEN, DMODEL, DMODEL);
  vtranspose_pack_kernel<<<vtC, 256, 0, stream>>>(pbuf, Vctb, SC_LEN);

  // fused dual attention + combine
  const unsigned attnBlocks = (unsigned)(NHEADS * (S_LEN / 16) / 4);
  attention_kernel<<<attnBlocks, 128, 0, stream>>>(Qb, Kb, Vtb, Kcb, Vctb, lam, out);
}
